// EmbeddedAttention_65154653880773
// MI455X (gfx1250) — matrix operand order repaired and checked by reading, not compiled, not run on hardware
//
#include <hip/hip_runtime.h>
#include <hip/hip_bf16.h>

// EmbeddedAttention for MI455X (gfx1250, wave32, WMMA).
// B=16, T=12, N=512, D=128, E=32.
// Pipeline:
//   k_prep_wv     : W_v -> W_v^T (bf16)                      [tiny]
//   k_qk_softmax  : qk = emb@W_qk+b (f32) ; q = rowsoftmax -> bf16 [T,N,D]
//   k_col_softmax : k  = colsoftmax(qk) stored transposed -> bf16 [T,N(m),D]
//   k_score       : score[t] = q@k    (WMMA bf16, f32 acc) -> bf16 [T,N,N]
//   k_vproj       : v = value@W_v+b   (WMMA bf16, f32 acc) -> bf16 transposed [B,T,D,N]
//   k_attn_out    : out = mask?score:-1e9 @ v  (WMMA bf16, f32 acc) -> f32 [B,T,N,D]

#define BB 16
#define TT 12
#define NNODE 512
#define DDIM 128
#define EDIM 32

typedef __attribute__((ext_vector_type(16))) __bf16 v16bf;
typedef __attribute__((ext_vector_type(8)))  __bf16 v8bf;
typedef __attribute__((ext_vector_type(4)))  __bf16 v4bf;
typedef __attribute__((ext_vector_type(8)))  float  v8f;

union AB16 { v16bf v; v8bf h[2]; __bf16 e[16]; };

__device__ __forceinline__ v8f wmma_bf16(v16bf a, v16bf b, v8f c) {
  return __builtin_amdgcn_wmma_f32_16x16x32_bf16(false, a, false, b, (short)0, c,
                                                 false, false);
}

// A operand (16x32, bf16, row-major src, ld in elements).
// Lane L: row = L&15, K base = (L<16?0:8): vgpr0..3 = K[base..base+7],
// vgpr4..7 = K[base+16..base+23] -> two contiguous 16B loads.
__device__ __forceinline__ v16bf load_a_bf(const __bf16* A, int ld, int row0,
                                           int k0, int lane) {
  int r = lane & 15, h = (lane >> 4) & 1;
  const __bf16* p = A + (size_t)(row0 + r) * ld + k0 + h * 8;
  AB16 u;
  u.h[0] = *(const v8bf*)(p);
  u.h[1] = *(const v8bf*)(p + 16);
  return u.v;
}

// B operand (32x16, bf16) from B^T stored row-major [Ncols x K].
// Lane L: col = L&15, K base = (L<16?0:8): K[base..base+7], K[base+16..base+23].
__device__ __forceinline__ v16bf load_b_bf(const __bf16* BT, int ld, int col0,
                                           int k0, int lane) {
  int r = lane & 15, h = (lane >> 4) & 1;
  const __bf16* p = BT + (size_t)(col0 + r) * ld + k0 + h * 8;
  AB16 u;
  u.h[0] = *(const v8bf*)(p);
  u.h[1] = *(const v8bf*)(p + 16);
  return u.v;
}

// A operand from fp32 source, converted to bf16 on the fly.
__device__ __forceinline__ v16bf load_a_f32(const float* A, int ld, int row0,
                                            int k0, int lane) {
  int r = lane & 15, h = (lane >> 4) & 1;
  const float* p = A + (size_t)(row0 + r) * ld + k0 + h * 8;
  float4 f0 = *(const float4*)(p);
  float4 f1 = *(const float4*)(p + 4);
  float4 f2 = *(const float4*)(p + 16);
  float4 f3 = *(const float4*)(p + 20);
  AB16 u;
  u.e[0]  = (__bf16)f0.x; u.e[1]  = (__bf16)f0.y; u.e[2]  = (__bf16)f0.z; u.e[3]  = (__bf16)f0.w;
  u.e[4]  = (__bf16)f1.x; u.e[5]  = (__bf16)f1.y; u.e[6]  = (__bf16)f1.z; u.e[7]  = (__bf16)f1.w;
  u.e[8]  = (__bf16)f2.x; u.e[9]  = (__bf16)f2.y; u.e[10] = (__bf16)f2.z; u.e[11] = (__bf16)f2.w;
  u.e[12] = (__bf16)f3.x; u.e[13] = (__bf16)f3.y; u.e[14] = (__bf16)f3.z; u.e[15] = (__bf16)f3.w;
  return u.v;
}

// ---------------- kernel 0: W_v -> W_v^T bf16 ----------------
__global__ void k_prep_wv(const float* __restrict__ Wv, __bf16* __restrict__ WvT) {
  int idx = blockIdx.x * blockDim.x + threadIdx.x;  // 128*128
  int d = idx >> 7, e = idx & 127;
  WvT[(size_t)d * DDIM + e] = (__bf16)Wv[(size_t)e * DDIM + d];
}

// ---------------- kernel 1: qk projection + row softmax (q) ----------------
// One wave per (t,n) row. Lane owns 4 of the 128 output features.
__global__ void k_qk_softmax(const float* __restrict__ emb,
                             const float* __restrict__ Wqk,
                             const float* __restrict__ bqk,
                             float* __restrict__ qk, __bf16* __restrict__ q_bf) {
  int lane = threadIdx.x & 31;
  int row = blockIdx.x * 8 + (threadIdx.x >> 5);  // < T*N = 6144
  float ev = emb[(size_t)row * EDIM + lane];      // one E element per lane
  int d0 = lane * 4;
  float4 acc = *(const float4*)(bqk + d0);
#pragma unroll
  for (int e = 0; e < EDIM; ++e) {
    float s = __shfl(ev, e, 32);
    float4 w = *(const float4*)(Wqk + (size_t)e * DDIM + d0);
    acc.x += s * w.x; acc.y += s * w.y; acc.z += s * w.z; acc.w += s * w.w;
  }
  *(float4*)(qk + (size_t)row * DDIM + d0) = acc;  // logits for column softmax
  // softmax over D within the wave
  float mx = fmaxf(fmaxf(acc.x, acc.y), fmaxf(acc.z, acc.w));
#pragma unroll
  for (int o = 16; o; o >>= 1) mx = fmaxf(mx, __shfl_xor(mx, o, 32));
  float e0 = __expf(acc.x - mx), e1 = __expf(acc.y - mx);
  float e2 = __expf(acc.z - mx), e3 = __expf(acc.w - mx);
  float s = e0 + e1 + e2 + e3;
#pragma unroll
  for (int o = 16; o; o >>= 1) s += __shfl_xor(s, o, 32);
  float inv = 1.0f / s;
  v4bf ov = {(__bf16)(e0 * inv), (__bf16)(e1 * inv), (__bf16)(e2 * inv),
             (__bf16)(e3 * inv)};
  *(v4bf*)(q_bf + (size_t)row * DDIM + d0) = ov;
}

// ---------------- kernel 2: column softmax over N -> k^T bf16 [T,N(m),D] ----
__global__ void k_col_softmax(const float* __restrict__ qk,
                              __bf16* __restrict__ kT_bf) {
  int t = blockIdx.x;        // 12
  int d = threadIdx.x;       // 128
  const float* base = qk + (size_t)t * NNODE * DDIM + d;
  float mx = -3.0e38f;
  for (int n = 0; n < NNODE; ++n) mx = fmaxf(mx, base[(size_t)n * DDIM]);
  float s = 0.f;
  for (int n = 0; n < NNODE; ++n) s += __expf(base[(size_t)n * DDIM] - mx);
  float inv = 1.0f / s;
  __bf16* ob = kT_bf + (size_t)t * NNODE * DDIM + d;
  for (int n = 0; n < NNODE; ++n)
    ob[(size_t)n * DDIM] = (__bf16)(__expf(base[(size_t)n * DDIM] - mx) * inv);
}

// ---------------- kernel 3: score[t] = q @ k  (N x N, K=128) ----------------
// Wave computes a 16(n) x 64(m) strip: A frag reused over 4 accumulators.
__global__ void k_score(const __bf16* __restrict__ q_bf,
                        const __bf16* __restrict__ kT_bf,
                        __bf16* __restrict__ score_bf) {
  int lane = threadIdx.x & 31;
  int gid = blockIdx.x * 8 + (threadIdx.x >> 5);  // < 12*256
  int t = gid >> 8;
  int rtile = gid & 255;
  int row0 = (rtile >> 3) * 16;
  int col0 = (rtile & 7) * 64;
  const __bf16* Q = q_bf + (size_t)t * NNODE * DDIM;
  const __bf16* K = kT_bf + (size_t)t * NNODE * DDIM;
  v8f acc[4] = {};
#pragma unroll
  for (int kk = 0; kk < DDIM; kk += 32) {
    v16bf a = load_a_bf(Q, DDIM, row0, kk, lane);
#pragma unroll
    for (int c = 0; c < 4; ++c) {
      v16bf b = load_b_bf(K, DDIM, col0 + c * 16, kk, lane);
      acc[c] = wmma_bf16(a, b, acc[c]);
    }
  }
  int r = lane & 15, h = (lane >> 4) & 1;
  __bf16* S = score_bf + (size_t)t * NNODE * NNODE;
#pragma unroll
  for (int c = 0; c < 4; ++c)
#pragma unroll
    for (int i = 0; i < 8; ++i)
      S[(size_t)(row0 + h * 8 + i) * NNODE + col0 + c * 16 + r] =
          (__bf16)acc[c][i];
}

// ---------------- kernel 4: v = value @ W_v + b_v, stored transposed -------
// Wave computes 16(m) x 128(d): value row loaded once, 8 accumulators.
__global__ void k_vproj(const float* __restrict__ value,
                        const __bf16* __restrict__ WvT,
                        const float* __restrict__ bv,
                        __bf16* __restrict__ v_bfT) {
  int lane = threadIdx.x & 31;
  int gid = blockIdx.x * 8 + (threadIdx.x >> 5);  // < 192*32
  int bt = gid >> 5;
  int row0 = (gid & 31) * 16;
  const float* V = value + (size_t)bt * NNODE * DDIM;
  v8f acc[8] = {};
#pragma unroll
  for (int kk = 0; kk < DDIM; kk += 32) {
    v16bf a = load_a_f32(V, DDIM, row0, kk, lane);
#pragma unroll
    for (int c = 0; c < 8; ++c) {
      v16bf b = load_b_bf(WvT, DDIM, c * 16, kk, lane);
      acc[c] = wmma_bf16(a, b, acc[c]);
    }
  }
  int r = lane & 15, h = (lane >> 4) & 1;
  __bf16* VT = v_bfT + (size_t)bt * DDIM * NNODE;  // [d][m]
#pragma unroll
  for (int c = 0; c < 8; ++c) {
    int d = c * 16 + r;
    float bias = bv[d];
#pragma unroll
    for (int i = 0; i < 8; ++i)
      VT[(size_t)d * NNODE + row0 + h * 8 + i] = (__bf16)(acc[c][i] + bias);
  }
}

// ---------------- kernel 5: out = (mask ? score : -1e9) @ v ----------------
// Wave computes 16(n) x 128(d); mask applied in the A-fragment load.
__global__ void k_attn_out(const __bf16* __restrict__ score_bf,
                           const int* __restrict__ mask,
                           const __bf16* __restrict__ v_bfT,
                           float* __restrict__ out) {
  int lane = threadIdx.x & 31;
  int gid = blockIdx.x * 8 + (threadIdx.x >> 5);  // < 192*32
  int bt = gid >> 5;
  int b = bt / TT, t = bt % TT;
  int row0 = (gid & 31) * 16;
  const __bf16* S = score_bf + (size_t)t * NNODE * NNODE;
  const int* M = mask + (size_t)b * NNODE * NNODE;
  const __bf16* VT = v_bfT + (size_t)bt * DDIM * NNODE;
  const __bf16 neg = (__bf16)(-1.0e9f);
  int r = lane & 15, h = (lane >> 4) & 1;
  v8f acc[8] = {};
  for (int k0 = 0; k0 < NNODE; k0 += 32) {
    // masked A fragment
    const __bf16* p = S + (size_t)(row0 + r) * NNODE + k0 + h * 8;
    const int* mp = M + (size_t)(row0 + r) * NNODE + k0 + h * 8;
    AB16 u;
    u.h[0] = *(const v8bf*)(p);
    u.h[1] = *(const v8bf*)(p + 16);
    int mv[16];
    *(int4*)(&mv[0])  = *(const int4*)(mp);
    *(int4*)(&mv[4])  = *(const int4*)(mp + 4);
    *(int4*)(&mv[8])  = *(const int4*)(mp + 16);
    *(int4*)(&mv[12]) = *(const int4*)(mp + 20);
#pragma unroll
    for (int j = 0; j < 16; ++j)
      if (mv[j] == 0) u.e[j] = neg;
    if (k0 + 32 < NNODE) __builtin_prefetch(p + 32, 0, 0);  // global_prefetch_b8
#pragma unroll
    for (int c = 0; c < 8; ++c) {
      v16bf bfr = load_b_bf(VT, NNODE, c * 16, k0, lane);
      acc[c] = wmma_bf16(u.v, bfr, acc[c]);
    }
  }
  float* O = out + (size_t)bt * NNODE * DDIM;
#pragma unroll
  for (int c = 0; c < 8; ++c)
#pragma unroll
    for (int i = 0; i < 8; ++i)
      O[(size_t)(row0 + h * 8 + i) * DDIM + c * 16 + r] = acc[c][i];
}

extern "C" void kernel_launch(void* const* d_in, const int* in_sizes, int n_in,
                              void* d_out, int out_size, void* d_ws,
                              size_t ws_size, hipStream_t stream) {
  const float* value = (const float*)d_in[0];   // [B,T,N,D]
  const float* emb   = (const float*)d_in[1];   // [T,N,E]
  const int*   amask = (const int*)d_in[2];     // [B,N,N]
  const float* Wqk   = (const float*)d_in[3];   // [E,D]
  const float* bqk   = (const float*)d_in[4];   // [D]
  const float* Wv    = (const float*)d_in[5];   // [D,D]
  const float* bv    = (const float*)d_in[6];   // [D]
  float* out = (float*)d_out;                   // [B,T,N,D]

  // workspace layout (256B aligned regions)
  char* w = (char*)d_ws;
  size_t off = 0;
  auto carve = [&](size_t bytes) {
    void* p = w + off;
    off = (off + bytes + 255) & ~(size_t)255;
    return p;
  };
  float*  qk_ws    = (float*)carve((size_t)TT * NNODE * DDIM * 4);        // 3 MB
  __bf16* q_bf     = (__bf16*)carve((size_t)TT * NNODE * DDIM * 2);       // 1.5 MB
  __bf16* kT_bf    = (__bf16*)carve((size_t)TT * NNODE * DDIM * 2);       // 1.5 MB
  __bf16* score_bf = (__bf16*)carve((size_t)TT * NNODE * NNODE * 2);      // 6 MB
  __bf16* WvT_bf   = (__bf16*)carve((size_t)DDIM * DDIM * 2);             // 32 KB
  __bf16* v_bfT    = (__bf16*)carve((size_t)BB * TT * DDIM * NNODE * 2);  // 24 MB

  k_prep_wv<<<(DDIM * DDIM) / 256, 256, 0, stream>>>(Wv, WvT_bf);
  k_qk_softmax<<<(TT * NNODE) / 8, 256, 0, stream>>>(emb, Wqk, bqk, qk_ws, q_bf);
  k_col_softmax<<<TT, DDIM, 0, stream>>>(qk_ws, kT_bf);
  k_score<<<(TT * 256) / 8, 256, 0, stream>>>(q_bf, kT_bf, score_bf);
  k_vproj<<<(BB * TT * 32) / 8, 256, 0, stream>>>(value, WvT_bf, bv, v_bfT);
  k_attn_out<<<(BB * TT * 32) / 8, 256, 0, stream>>>(score_bf, amask, v_bfT, out);
}
